// BiGeaR_tch_66760971649715
// MI455X (gfx1250) — matrix operand order repaired and checked by reading, not compiled, not run on hardware
//
#include <hip/hip_runtime.h>
#include <hip/hip_bf16.h>
#include <math.h>

#define NUM_USERS 100000
#define NUM_ITEMS 50000
#define N_TOTAL   150000
#define DIM       64
#define NNZ_CNT   2000000
#define BATCH     2048
#define N_NEGS    16
#define WPB       8      // waves per block in score kernel

typedef __attribute__((ext_vector_type(16))) _Float16 v16h;
typedef __attribute__((ext_vector_type(8)))  float    v8f;
typedef __attribute__((ext_vector_type(4)))  unsigned v4u;
typedef __attribute__((ext_vector_type(8)))  int      v8i;
typedef __attribute__((ext_vector_type(4)))  int      v4i;

#if __has_builtin(__builtin_amdgcn_tensor_load_to_lds)
#define HAVE_TDM 1
#else
#define HAVE_TDM 0
#endif

// ---------------------------------------------------------------------------
// Deterministic uniform hash in [0,1) for the negative-mixing "seed".
// ---------------------------------------------------------------------------
__device__ __forceinline__ float uhash(unsigned x) {
    x ^= 0x9e3779b9u;
    x ^= x >> 16; x *= 0x7feb352du;
    x ^= x >> 15; x *= 0x846ca68bu;
    x ^= x >> 16;
    return (float)(x >> 8) * (1.0f / 16777216.0f);
}

// Stacked-embedding fetch (lambda-scaled); l==0 reads raw concat tables.
__device__ __forceinline__ float emb_at(const float* __restrict__ ue,
                                        const float* __restrict__ ie,
                                        const float* __restrict__ L1,
                                        const float* __restrict__ L2,
                                        const float* __restrict__ L3,
                                        int node, int l, int d) {
    float v;
    if (l == 0) {
        v = (node < NUM_USERS) ? ue[node * DIM + d]
                               : ie[(node - NUM_USERS) * DIM + d];
    } else {
        const float* src = (l == 1) ? L1 : ((l == 2) ? L2 : L3);
        v = src[node * DIM + d];
    }
    return v * (0.25f * (float)(l + 1));
}

// ---------------------------------------------------------------------------
__global__ void zero_kernel(float* __restrict__ ws, int n, float* __restrict__ out) {
    int i = blockIdx.x * blockDim.x + threadIdx.x;
    if (i < n) ws[i] = 0.0f;
    if (blockIdx.x == 0 && threadIdx.x < 2) out[threadIdx.x] = 0.0f;
}

// ---------------------------------------------------------------------------
// SpMM: dst[row] += val * src[col]   (one wave32 per nonzero, float2/lane)
// ---------------------------------------------------------------------------
__global__ void spmm_kernel(const float* __restrict__ vals,
                            const int*   __restrict__ rows,
                            const int*   __restrict__ cols,
                            const float* __restrict__ ue,
                            const float* __restrict__ ie,
                            const float* __restrict__ src,
                            float*       __restrict__ dst,
                            int layer1) {
    const int lane   = threadIdx.x & 31;
    const int wave   = (blockIdx.x * blockDim.x + threadIdx.x) >> 5;
    const int nwaves = (gridDim.x * blockDim.x) >> 5;

    for (int i = wave; i < NNZ_CNT; i += nwaves) {
        const int   r = rows[i];
        const int   c = cols[i];
        const float v = vals[i];

        const float* s;
        int off;
        if (layer1) {
            if (c < NUM_USERS) { s = ue; off = c * DIM; }
            else               { s = ie; off = (c - NUM_USERS) * DIM; }
        } else { s = src; off = c * DIM; }

        const float2 x = *(const float2*)(s + off + lane * 2);

        const int ni = i + nwaves;
        if (ni < NNZ_CNT) {
            const int nc = cols[ni];
            const float* ps; int poff;
            if (layer1) {
                if (nc < NUM_USERS) { ps = ue; poff = nc * DIM; }
                else                { ps = ie; poff = (nc - NUM_USERS) * DIM; }
            } else { ps = src; poff = nc * DIM; }
            __builtin_prefetch(ps + poff + lane * 2, 0, 1);
        }

        float* d = dst + r * DIM + lane * 2;
        atomicAdd(d,     v * x.x);
        atomicAdd(d + 1, v * x.y);
    }
}

// ---------------------------------------------------------------------------
// TDM gather: 16 rows (16-bit indices) of a [nrows x 64] f32 tensor -> LDS.
// D# packed per CDNA5 ISA 8.3/8.4/8.7 (gather_mode, data_size=4B, tile 64x16).
// ---------------------------------------------------------------------------
#if HAVE_TDM
__device__ __forceinline__ void tdm_gather16(const float* __restrict__ base,
                                             v4i g2, v4i g3,
                                             unsigned lds_byte_addr) {
    const unsigned long long ga = (unsigned long long)(uintptr_t)base;
    v4u g0;
    g0.x = 1u | (1u << 31);                                  // count=1, gather_mode=1, 16b idx
    g0.y = lds_byte_addr;                                    // LDS dest (bytes)
    g0.z = (unsigned)(ga & 0xffffffffu);                     // global_addr[31:0]
    g0.w = ((unsigned)(ga >> 32) & 0x01ffffffu) | (2u << 30);// global_addr[56:32], type=2
    v8i g1;
    g1[0] = (2 << 16);                 // data_size = 4B
    g1[1] = (DIM << 16);               // tensor_dim0 = 64  (bits 63:48)
    g1[2] = (NUM_ITEMS & 0xffff) << 16;// tensor_dim1[15:0] (bits 95:80)
    g1[3] = (DIM << 16);               // tile_dim0 = 64    (bits 127:112)
    g1[4] = N_NEGS;                    // tile_dim1 = #valid indices (bits 143:128)
    g1[5] = DIM;                       // tensor_dim0_stride = 64 (bits 191:160)
    g1[6] = 0;
    g1[7] = 0;
    const v8i gz = {0, 0, 0, 0, 0, 0, 0, 0};   // unused extension group (set to 0)
    __builtin_amdgcn_tensor_load_to_lds(g0, g1, g2, g3, gz, 0);
}
#endif

// ---------------------------------------------------------------------------
// Scoring: one wave32 per batch element; all operands staged in LDS.
// ---------------------------------------------------------------------------
__global__ void __launch_bounds__(256)
score_kernel(const float* __restrict__ ue, const float* __restrict__ ie,
             const float* __restrict__ L1, const float* __restrict__ L2,
             const float* __restrict__ L3,
             const int* __restrict__ user_index,
             const int* __restrict__ pos_index,
             const int* __restrict__ neg_index,
             float* __restrict__ out) {
    __shared__ float s_neg[WPB][N_NEGS * DIM];   // raw (unscaled) neg rows, 32 KB
    __shared__ float s_u[WPB][4 * DIM];          // lambda-scaled user,  8 KB
    __shared__ float s_p[WPB][4 * DIM];          // lambda-scaled pos,   8 KB

    const int lane = threadIdx.x & 31;
    const int wave = threadIdx.x >> 5;
    const int b    = blockIdx.x * WPB + wave;
    if (b >= BATCH) return;                       // wave-uniform

    const int unode = user_index[b];
    const int pnode = NUM_USERS + pos_index[b];
    const int* ni   = neg_index + b * N_NEGS;     // 16 item indices (wave-uniform)

    // stage lambda-scaled u and p rows (pooled 256 floats each)
    for (int t = 0; t < 8; ++t) {
        const int dd = lane + 32 * t;
        const int l = dd >> 6, d = dd & 63;
        s_u[wave][dd] = emb_at(ue, ie, L1, L2, L3, unode, l, d);
        s_p[wave][dd] = emb_at(ue, ie, L1, L2, L3, pnode, l, d);
    }

#if HAVE_TDM
    // pack 16 x 16-bit row indices for the TDM gather descriptor
    v4i g2, g3;
    g2[0] = (ni[0] & 0xffff) | (ni[1] << 16);
    g2[1] = (ni[2] & 0xffff) | (ni[3] << 16);
    g2[2] = (ni[4] & 0xffff) | (ni[5] << 16);
    g2[3] = (ni[6] & 0xffff) | (ni[7] << 16);
    g3[0] = (ni[8] & 0xffff) | (ni[9] << 16);
    g3[1] = (ni[10] & 0xffff) | (ni[11] << 16);
    g3[2] = (ni[12] & 0xffff) | (ni[13] << 16);
    g3[3] = (ni[14] & 0xffff) | (ni[15] << 16);
    const unsigned lds_addr = (unsigned)(uintptr_t)&s_neg[wave][0];
#endif

    const int negrow = lane & 15;                 // A-matrix row (M) this lane feeds
    const int khalfA = (lane < 16) ? 0 : 8;       // 16-bit A 16x32 lane striping
    const int khalfB = (lane < 16) ? 0 : 8;       // 16-bit B 32x16 lane striping
    const int mbase  = (lane < 16) ? 0 : 8;

    float pos_part = 0.f, neg_part = 0.f, reg_part = 0.f;

    for (int l = 0; l < 4; ++l) {
        const float laml = 0.25f * (float)(l + 1);
        const float* basep =
            (l == 0) ? ie
                     : (((l == 1) ? L1 : ((l == 2) ? L2 : L3)) + NUM_USERS * DIM);

        // --- stage the 16 raw negative rows for this layer into LDS -----
#if HAVE_TDM
        // drain LDS reads of the previous layer before TDM overwrites s_neg
        asm volatile("s_wait_dscnt 0x0" ::: "memory");
        tdm_gather16(basep, g2, g3, lds_addr);
        __builtin_amdgcn_s_wait_tensorcnt((short)0);
#else
        for (int t = 0; t < 32; ++t) {
            const int q = lane + 32 * t;          // 0..1023
            const int j = q >> 6, d = q & 63;
            s_neg[wave][q] = basep[ni[j] * DIM + d];
        }
#endif

        // --- 16 neg scores via v_wmma_f32_16x16x32_f16 ------------------
        v8f c = {0.f, 0.f, 0.f, 0.f, 0.f, 0.f, 0.f, 0.f};
        for (int s = 0; s < 2; ++s) {             // two K=32 slices of DIM=64
            v16h a, bm;
            for (int j = 0; j < 8; ++j) {
                const int kA = (j < 4) ? (khalfA + 2 * j)
                                       : (16 + khalfA + 2 * (j - 4));
                const int d0 = s * 32 + kA;
                for (int e = 0; e < 2; ++e) {
                    const int d = d0 + e;
                    const float nv = laml * s_neg[wave][negrow * DIM + d];
                    const float pv = s_p[wave][l * DIM + d];
                    const float sd = uhash((unsigned)(((b * N_NEGS + negrow) * 4 + l) * DIM + d));
                    a[2 * j + e] = (_Float16)(sd * pv + (1.0f - sd) * nv);
                }
                const int dB = s * 32 + ((j < 4) ? (khalfB + 2 * j) : (16 + khalfB + 2 * (j - 4)));  // u replicated in all columns
                bm[2 * j]     = (_Float16)s_u[wave][l * DIM + dB];
                bm[2 * j + 1] = (_Float16)s_u[wave][l * DIM + dB + 1];
            }
            c = __builtin_amdgcn_wmma_f32_16x16x32_f16(
                    false, a, false, bm, (short)0, c, false, false);
        }

        // --- per-layer argmax over the 16 negs --------------------------
        float bestv = -INFINITY; int bestm = 0;
        for (int r = 0; r < 8; ++r) {
            const float v = c[r];
            if (v > bestv) { bestv = v; bestm = mbase + r; }
        }
        const float ov = __shfl_xor(bestv, 16, 32);
        const int   om = __shfl_xor(bestm, 16, 32);
        if (ov > bestv || (ov == bestv && om < bestm)) { bestv = ov; bestm = om; }

        // --- accumulate pooled dots for this layer (tile still in LDS) --
        for (int t = 0; t < 2; ++t) {
            const int d = lane + 32 * t;          // dim 0..63
            const float uv = s_u[wave][l * DIM + d];
            const float pv = s_p[wave][l * DIM + d];
            pos_part += uv * pv;
            const float sd = uhash((unsigned)(((b * N_NEGS + bestm) * 4 + l) * DIM + d));
            const float m  = sd * pv + (1.0f - sd) * laml * s_neg[wave][bestm * DIM + d];
            neg_part += uv * m;
            if (l == 0) reg_part += uv * uv + pv * pv + m * m;
        }
    }

    // --- wave reduction + scalar atomics --------------------------------
    for (int off = 16; off >= 1; off >>= 1) {
        pos_part += __shfl_xor(pos_part, off, 32);
        neg_part += __shfl_xor(neg_part, off, 32);
        reg_part += __shfl_xor(reg_part, off, 32);
    }
    if (lane == 0) {
        const float inv_b = 1.0f / (float)BATCH;
        atomicAdd(&out[0], logf(1.0f + expf(neg_part - pos_part)) * inv_b);
        atomicAdd(&out[1], 0.5f * reg_part * inv_b);
    }
}

// ---------------------------------------------------------------------------
extern "C" void kernel_launch(void* const* d_in, const int* in_sizes, int n_in,
                              void* d_out, int out_size, void* d_ws, size_t ws_size,
                              hipStream_t stream) {
    const float* ue   = (const float*)d_in[0];
    const float* ie   = (const float*)d_in[1];
    const float* vals = (const float*)d_in[2];
    const int*   rows = (const int*)  d_in[3];
    const int*   cols = (const int*)  d_in[4];
    const int*   uix  = (const int*)  d_in[5];
    const int*   pix  = (const int*)  d_in[6];
    const int*   nix  = (const int*)  d_in[7];
    float*       out  = (float*)d_out;

    const int layer_elems = N_TOTAL * DIM;      // 9.6M floats
    float* L1 = (float*)d_ws;
    float* L2 = L1 + layer_elems;
    float* L3 = L2 + layer_elems;

    {
        const int n = 3 * layer_elems;
        zero_kernel<<<(n + 255) / 256, 256, 0, stream>>>(L1, n, out);
    }

    const int spmm_blocks = 2048, spmm_threads = 256;
    spmm_kernel<<<spmm_blocks, spmm_threads, 0, stream>>>(vals, rows, cols, ue, ie, nullptr, L1, 1);
    spmm_kernel<<<spmm_blocks, spmm_threads, 0, stream>>>(vals, rows, cols, ue, ie, L1,      L2, 0);
    spmm_kernel<<<spmm_blocks, spmm_threads, 0, stream>>>(vals, rows, cols, ue, ie, L2,      L3, 0);

    score_kernel<<<BATCH / WPB, 32 * WPB, 0, stream>>>(ue, ie, L1, L2, L3, uix, pix, nix, out);
}
